// RNN_8761733283938
// MI455X (gfx1250) — matrix operand order repaired and checked by reading, not compiled, not run on hardware
//
#include <hip/hip_runtime.h>

// RNN: X[64,1024,256], h0[64,256], Wx[256,256], Wh[256,256], b[256]
// out[64,1024,256] f32.
#define T_S 1024

typedef __attribute__((ext_vector_type(16))) __bf16 v16bf;
typedef __attribute__((ext_vector_type(16))) float  v16f;
typedef __attribute__((ext_vector_type(8)))  float  v8f;
typedef __attribute__((ext_vector_type(4)))  float  f32x4;
typedef __attribute__((ext_vector_type(4)))  unsigned int u32x4;

union FragU {
  u32x4 q[2];
  v16bf v;
};

// Hardware tanh if the toolchain exposes it (gfx1250 TRANS op); else libm.
#if __has_builtin(__builtin_amdgcn_tanhf)
__device__ __forceinline__ float fast_tanh(float x) { return __builtin_amdgcn_tanhf(x); }
#elif __has_builtin(__builtin_amdgcn_tanh_f32)
__device__ __forceinline__ float fast_tanh(float x) { return __builtin_amdgcn_tanh_f32(x); }
#else
__device__ __forceinline__ float fast_tanh(float x) { return tanhf(x); }
#endif

// ---- prep: Wx,Wh -> column-major ("B-friendly") bf16 copies in workspace ----
__global__ __launch_bounds__(256) void prep_transpose(
    const float* __restrict__ Wx, const float* __restrict__ Wh,
    __bf16* __restrict__ WxT, __bf16* __restrict__ WhT) {
  int idx = blockIdx.x * 256 + threadIdx.x;  // 0..65535
  int k = idx >> 8;
  int n = idx & 255;
  WxT[n * 256 + k] = (__bf16)Wx[idx];
  WhT[n * 256 + k] = (__bf16)Wh[idx];
}

// ---- phase 1: Xp = X @ Wx + b  (written into d_out, overwritten by scan) ----
// grid (512,4), block 256. Each wave: 16 rows x 64 cols (4 WMMA tiles).
__global__ __launch_bounds__(256) void xp_gemm(
    const float* __restrict__ X, const __bf16* __restrict__ WxT,
    const float* __restrict__ bias, float* __restrict__ out) {
  const int lane  = threadIdx.x & 31;
  const int wave  = threadIdx.x >> 5;
  const int mlane = lane & 15;
  const int half  = lane >> 4;
  const int row0  = (blockIdx.x * 8 + wave) * 16;  // flat (n*T+t) row
  const int n0    = blockIdx.y * 64;

  v8f acc[4] = {};
  const float* arow = X + (size_t)(row0 + mlane) * 256;

#pragma unroll
  for (int ks = 0; ks < 8; ++ks) {
    const int kbA = ks * 32 + half * 8;
    f32x4 x0 = *(const f32x4*)(arow + kbA);
    f32x4 x1 = *(const f32x4*)(arow + kbA + 4);
    f32x4 x2 = *(const f32x4*)(arow + kbA + 16);
    f32x4 x3 = *(const f32x4*)(arow + kbA + 20);
    v16f xf;
#pragma unroll
    for (int e = 0; e < 4; ++e) {
      xf[e]      = x0[e];
      xf[4 + e]  = x1[e];
      xf[8 + e]  = x2[e];
      xf[12 + e] = x3[e];
    }
    const v16bf av = __builtin_convertvector(xf, v16bf);  // packed v_cvt
    const int kbB = ks * 32 + half * 8;
#pragma unroll
    for (int tile = 0; tile < 4; ++tile) {
      const int n = n0 + tile * 16 + mlane;
      FragU ub;
      const u32x4* pb = (const u32x4*)(WxT + (size_t)n * 256 + kbB);
      ub.q[0] = pb[0];
      ub.q[1] = pb[2];
      acc[tile] = __builtin_amdgcn_wmma_f32_16x16x32_bf16(
          false, av, false, ub.v, (short)0, acc[tile], false, false);
    }
  }
#pragma unroll
  for (int tile = 0; tile < 4; ++tile) {
    const int n = n0 + tile * 16 + mlane;
    const float bb = bias[n];
#pragma unroll
    for (int v = 0; v < 8; ++v) {
      const int gm = row0 + v + half * 8;
      out[(size_t)gm * 256 + n] = acc[tile][v] + bb;
    }
  }
}

// ---- phase 2: sequential scan. 4 blocks x 16 batch rows, 8 waves/block. ----
// Wh fragments pinned in VGPRs for all 1024 steps; h double-buffered in LDS
// with a padded pitch (264) so the 16-row A-fragment loads are conflict-free.
#define HPAD 264

__global__ __launch_bounds__(256) void rnn_scan(
    const float* __restrict__ h0, const __bf16* __restrict__ WhT,
    float* out) {
  __shared__ __attribute__((aligned(16))) __bf16 hbuf[2][16][HPAD];

  const int lane  = threadIdx.x & 31;
  const int wave  = threadIdx.x >> 5;
  const int mlane = lane & 15;
  const int half  = lane >> 4;
  const int row0  = blockIdx.x * 16;  // batch rows [row0, row0+16)

  // h_0 -> hbuf[0] (bf16)
  for (int i = threadIdx.x; i < 16 * 256; i += 256) {
    int r = i >> 8, c = i & 255;
    hbuf[0][r][c] = (__bf16)h0[(size_t)(row0 + r) * 256 + c];
  }

  // Persistent B fragments: this wave's two 16-col tiles of Wh, all K.
  v16bf Bf[2][8];
#pragma unroll
  for (int tile = 0; tile < 2; ++tile) {
    const int n = wave * 32 + tile * 16 + mlane;
#pragma unroll
    for (int ks = 0; ks < 8; ++ks) {
      const int kb = ks * 32 + half * 8;
      FragU u;
      const u32x4* p = (const u32x4*)(WhT + (size_t)n * 256 + kb);
      u.q[0] = p[0];
      u.q[1] = p[2];
      Bf[tile][ks] = u.v;
    }
  }

  // Preload Xp(t=0) from `out` (holds Xp after xp_gemm).
  float xp_cur[16], xp_next[16];
#pragma unroll
  for (int tile = 0; tile < 2; ++tile)
#pragma unroll
    for (int v = 0; v < 8; ++v) {
      const int m = v + half * 8;
      const int n = wave * 32 + tile * 16 + mlane;
      xp_cur[tile * 8 + v] = out[(size_t)(row0 + m) * T_S * 256 + n];
    }

  __syncthreads();

  for (int t = 0; t < T_S; ++t) {
    // Software pipeline: pull Xp(t+1) while this step's WMMAs execute.
    if (t + 1 < T_S) {
#pragma unroll
      for (int tile = 0; tile < 2; ++tile)
#pragma unroll
        for (int v = 0; v < 8; ++v) {
          const int m = v + half * 8;
          const int n = wave * 32 + tile * 16 + mlane;
          xp_next[tile * 8 + v] =
              out[((size_t)(row0 + m) * T_S + (t + 1)) * 256 + n];
        }
      if (t + 2 < T_S) {  // warm caches for t+2 (global_prefetch_b8)
        __builtin_prefetch(
            &out[((size_t)(row0 + mlane) * T_S + (t + 2)) * 256 + wave * 32],
            0, 3);
      }
    }

    v8f acc[2] = {};
    const __bf16(*hcur)[HPAD] = hbuf[t & 1];
#pragma unroll
    for (int ks = 0; ks < 8; ++ks) {
      const int kb = ks * 32 + half * 8;
      FragU ua;
      ua.q[0] = *(const u32x4*)&hcur[mlane][kb];
      ua.q[1] = *(const u32x4*)&hcur[mlane][kb + 16];
      acc[0] = __builtin_amdgcn_wmma_f32_16x16x32_bf16(
          false, ua.v, false, Bf[0][ks], (short)0, acc[0], false, false);
      acc[1] = __builtin_amdgcn_wmma_f32_16x16x32_bf16(
          false, ua.v, false, Bf[1][ks], (short)0, acc[1], false, false);
    }

#pragma unroll
    for (int tile = 0; tile < 2; ++tile)
#pragma unroll
      for (int v = 0; v < 8; ++v) {
        const int m = v + half * 8;
        const int n = wave * 32 + tile * 16 + mlane;
        const float hv = fast_tanh(acc[tile][v] + xp_cur[tile * 8 + v]);
        out[((size_t)(row0 + m) * T_S + t) * 256 + n] = hv;   // hs output
        hbuf[(t + 1) & 1][m][n] = (__bf16)hv;                 // next-step A
      }

    __syncthreads();
#pragma unroll
    for (int i = 0; i < 16; ++i) xp_cur[i] = xp_next[i];
  }
}

extern "C" void kernel_launch(void* const* d_in, const int* in_sizes, int n_in,
                              void* d_out, int out_size, void* d_ws,
                              size_t ws_size, hipStream_t stream) {
  (void)in_sizes; (void)n_in; (void)out_size; (void)ws_size;
  const float* X  = (const float*)d_in[0];
  const float* h0 = (const float*)d_in[1];
  const float* Wx = (const float*)d_in[2];
  const float* Wh = (const float*)d_in[3];
  const float* b  = (const float*)d_in[4];
  float* out = (float*)d_out;

  __bf16* WxT = (__bf16*)d_ws;          // 128 KB
  __bf16* WhT = WxT + 256 * 256;        // 128 KB

  prep_transpose<<<256, 256, 0, stream>>>(Wx, Wh, WxT, WhT);
  dim3 g1(512, 4, 1);
  xp_gemm<<<g1, 256, 0, stream>>>(X, WxT, b, out);
  rnn_scan<<<4, 256, 0, stream>>>(h0, WhT, out);
}
